// CausalSelfAttentionWithMemory_80814104641601
// MI455X (gfx1250) — matrix operand order repaired and checked by reading, not compiled, not run on hardware
//
#include <hip/hip_runtime.h>

// ---------------------------------------------------------------------------
// CDNA5 (gfx1250) causal self-attention with memory tokens.
// bf16 WMMA (v_wmma_f32_16x16x32_bf16), fp32 accumulate, async-to-LDS
// double-buffered GEMM staging (ASYNCcnt), flash attention w/ LDS transpose.
// Pipeline: cvt(x,W) -> QKV GEMM -> flash attention -> proj GEMM.
// ---------------------------------------------------------------------------

typedef __attribute__((ext_vector_type(16))) __bf16 v16bf;
typedef __attribute__((ext_vector_type(8)))  float  v8f;
typedef __attribute__((ext_vector_type(4)))  int    v4i;

typedef __attribute__((address_space(1))) v4i gv4i_t;  // global int4
typedef __attribute__((address_space(3))) v4i lv4i_t;  // LDS int4

union AF {           // one 16x32 bf16 A/B fragment = 8 VGPRs
  v16bf v;
  uint4 q[2];
};

static __device__ __forceinline__ v8f wmma_bf16(v16bf a, v16bf b, v8f c) {
  // D = A(16x32) * B(32x16) + C, fp32 accumulate
  return __builtin_amdgcn_wmma_f32_16x16x32_bf16(false, a, false, b, (short)0, c,
                                                 false, false);
}

// 16-byte per-lane async DMA: global -> LDS (tracked by ASYNCcnt)
static __device__ __forceinline__ void async_cp16(const __bf16* g, __bf16* l) {
#if __has_builtin(__builtin_amdgcn_global_load_async_to_lds_b128)
  __builtin_amdgcn_global_load_async_to_lds_b128((gv4i_t*)g, (lv4i_t*)l, 0, 0);
#else
  asm volatile("global_load_async_to_lds_b128 %0, %1, off"
               :: "v"((unsigned)(unsigned long long)l),
                  "v"((unsigned long long)g)
               : "memory");
#endif
}

#if __has_builtin(__builtin_amdgcn_s_wait_asynccnt)
#define WAIT_ASYNC(n) __builtin_amdgcn_s_wait_asynccnt(n)
#else
#define WAIT_ASYNC(n) asm volatile("s_wait_asynccnt %0" :: "i"(n) : "memory")
#endif

static __device__ __forceinline__ unsigned pack2(float a, float b) {
  union { __bf16 h; unsigned short u; } x, y;
  x.h = (__bf16)a; y.h = (__bf16)b;
  return (unsigned)x.u | ((unsigned)y.u << 16);
}

static __device__ __forceinline__ v8f vzero8() {
  v8f z;
#pragma unroll
  for (int i = 0; i < 8; ++i) z[i] = 0.0f;
  return z;
}

#define BATCH 4
#define SEQL  2056     // L = 8 memory + 2048 sequence
#define NMEM  8
#define NH    16
#define HD    64
#define CDIM  1024
#define NTOK  (BATCH * SEQL)   // 8224
#define KTILES 65              // ceil(2056/32)

// GEMM tiling
#define TM 128
#define TN 128
#define TKK 32
#define NKSTEP (CDIM / TKK)    // 32
#define ASTR 40    // LDS A row stride in halfwords (conflict-free 16B frag reads)
#define BSTR 136   // LDS B row stride in halfwords

// ---------------------------------------------------------------------------
// fp32 -> bf16 conversion (vectorized x4)
// ---------------------------------------------------------------------------
__global__ void cvt_kernel(const float* __restrict__ s, __bf16* __restrict__ d, int n4) {
  int i = blockIdx.x * blockDim.x + threadIdx.x;
  if (i < n4) {
    float4 f = ((const float4*)s)[i];
    uint2 u;
    u.x = pack2(f.x, f.y);
    u.y = pack2(f.z, f.w);
    ((uint2*)d)[i] = u;
  }
}

// ---------------------------------------------------------------------------
// QKV GEMM: xb(8224x1024 bf16) @ W_attn(1024x3072 bf16) ->
//   Q (B,H,L,hd) bf16, KT (B,H,hd,L) bf16 (transposed!), V (B,H,L,hd) bf16
// 256 thr = 8 waves; block tile 128x128; wave tile 64x32; K step 32.
// Double-buffered async-to-LDS tile staging.
// ---------------------------------------------------------------------------
__global__ __launch_bounds__(256) void qkv_gemm_kernel(
    const __bf16* __restrict__ xb, const __bf16* __restrict__ w,
    __bf16* __restrict__ Q, __bf16* __restrict__ KT, __bf16* __restrict__ V) {
  __shared__ __bf16 As[2][TM * ASTR];
  __shared__ __bf16 Bs[2][TKK * BSTR];

  const int tid  = threadIdx.x;
  const int lane = tid & 31;
  const int wave = tid >> 5;
  const int wm   = wave >> 2;      // 0..1
  const int wn   = wave & 3;       // 0..3
  const int half = lane >> 4;
  const int l15  = lane & 15;
  const int mb   = blockIdx.x * TM;
  const int nb   = blockIdx.y * TN;

  // async chunk mapping: A tile 128x32 (512 x 16B chunks), B tile 32x128
  const int ar0 = tid >> 2;            // A rows: tid>>2 and +64
  const int ac0 = (tid & 3) * 8;
  const int br0 = tid >> 4;            // B rows: tid>>4 and +16
  const int bc0 = (tid & 15) * 8;

  const __bf16* aA0 = xb + (size_t)min(mb + ar0, NTOK - 1) * CDIM + ac0;
  const __bf16* aA1 = xb + (size_t)min(mb + ar0 + 64, NTOK - 1) * CDIM + ac0;
  const __bf16* aB0 = w + (size_t)br0 * (3 * CDIM) + nb + bc0;
  const __bf16* aB1 = w + (size_t)(br0 + 16) * (3 * CDIM) + nb + bc0;

  auto issue = [&](int kk, int bufi) {
    async_cp16(aA0 + kk, &As[bufi][ar0 * ASTR + ac0]);
    async_cp16(aA1 + kk, &As[bufi][(ar0 + 64) * ASTR + ac0]);
    async_cp16(aB0 + (size_t)kk * (3 * CDIM), &Bs[bufi][br0 * BSTR + bc0]);
    async_cp16(aB1 + (size_t)kk * (3 * CDIM), &Bs[bufi][(br0 + 16) * BSTR + bc0]);
  };

  v8f acc[4][2];
#pragma unroll
  for (int mi = 0; mi < 4; ++mi)
#pragma unroll
    for (int ni = 0; ni < 2; ++ni) acc[mi][ni] = vzero8();

  issue(0, 0);
  for (int ks = 0; ks < NKSTEP; ++ks) {
    const int cur = ks & 1;
    if (ks + 1 < NKSTEP) {
      issue((ks + 1) * TKK, cur ^ 1);
      WAIT_ASYNC(4);   // in-order: first 4 outstanding (cur buffer) are done
    } else {
      WAIT_ASYNC(0);
    }
    __syncthreads();

    AF a[4], b[2];
#pragma unroll
    for (int mi = 0; mi < 4; ++mi) {
      int row = wm * 64 + mi * 16 + l15;
      a[mi].q[0] = *(const uint4*)&As[cur][row * ASTR + half * 8];
      a[mi].q[1] = *(const uint4*)&As[cur][row * ASTR + 16 + half * 8];
    }
#pragma unroll
    for (int ni = 0; ni < 2; ++ni) {
      int col = wn * 32 + ni * 16;
#pragma unroll
      for (int s = 0; s < 16; ++s) {
        int k = (s < 8) ? (half * 8 + s) : (16 + half * 8 + (s - 8));
        b[ni].v[s] = Bs[cur][k * BSTR + col + l15];
      }
    }
#pragma unroll
    for (int mi = 0; mi < 4; ++mi)
#pragma unroll
      for (int ni = 0; ni < 2; ++ni)
        acc[mi][ni] = wmma_bf16(a[mi].v, b[ni].v, acc[mi][ni]);
    __syncthreads();   // protect cur buffer before it is re-filled at ks+1
  }

  // --- epilogue: scatter to Q / K^T / V bf16 ---
#pragma unroll
  for (int mi = 0; mi < 4; ++mi) {
#pragma unroll
    for (int ni = 0; ni < 2; ++ni) {
#pragma unroll
      for (int r = 0; r < 8; ++r) {
        int grow = mb + wm * 64 + mi * 16 + r + 8 * half;
        int gcol = nb + wn * 32 + ni * 16 + l15;
        if (grow < NTOK) {
          __bf16 bv = (__bf16)acc[mi][ni][r];
          int region = gcol >> 10;         // 0=Q 1=K 2=V
          int cc = gcol & (CDIM - 1);
          int h = cc >> 6, d = cc & (HD - 1);
          int bI = grow / SEQL, li = grow % SEQL;
          int bh = bI * NH + h;
          if (region == 0)      Q[((size_t)bh * SEQL + li) * HD + d] = bv;
          else if (region == 1) KT[((size_t)bh * HD + d) * SEQL + li] = bv;
          else                  V[((size_t)bh * SEQL + li) * HD + d] = bv;
        }
      }
    }
  }
}

// ---------------------------------------------------------------------------
// Flash attention. Grid: (33 q-tiles of 64, 64 b*h). Block: 128 thr = 4 waves.
// Each wave owns 16 query rows; streams 32-key tiles with online softmax.
// Mask: allowed(q,k) = (q < NMEM) || (k <= q); plus k < L.
// ---------------------------------------------------------------------------
__global__ __launch_bounds__(128) void attn_kernel(
    const __bf16* __restrict__ Q, const __bf16* __restrict__ KT,
    const __bf16* __restrict__ V, __bf16* __restrict__ Y) {
  const int lane = threadIdx.x & 31;
  const int wave = threadIdx.x >> 5;
  const int half = lane >> 4;
  const int l15  = lane & 15;
  const int bh   = blockIdx.y;              // b*NH + h
  const int qb   = blockIdx.x;
  const int q0   = qb * 64 + wave * 16;

  __shared__ __bf16 pbuf[4][16 * 40];       // per-wave P transpose staging
  __bf16* pb = pbuf[wave];

  AF qf[2];
  {
    int qrow = min(q0 + l15, SEQL - 1);
    const __bf16* qp = Q + ((size_t)bh * SEQL + qrow) * HD;
#pragma unroll
    for (int c = 0; c < 2; ++c) {
      qf[c].q[0] = *(const uint4*)(qp + c * 32 + half * 8);
      qf[c].q[1] = *(const uint4*)(qp + c * 32 + 16 + half * 8);
    }
  }

  v8f o[4];
#pragma unroll
  for (int nf = 0; nf < 4; ++nf) o[nf] = vzero8();
  float m[8], lsum[8];
#pragma unroll
  for (int r = 0; r < 8; ++r) { m[r] = -1.0e30f; lsum[r] = 0.0f; }

  const float sscale = 0.125f * 1.4426950408889634f;  // 1/sqrt(64) * log2(e)
  const __bf16* ktbase = KT + (size_t)bh * HD * SEQL;
  const __bf16* vbase  = V + (size_t)bh * SEQL * HD;

  const int ktiles = (qb == 0) ? KTILES : min(2 * qb + 2, KTILES);

  for (int kt = 0; kt < ktiles; ++kt) {
    const int kbase = kt * 32;

    // ---- S = Q * K^T for 32 keys (two 16-key subtiles) ----
    v8f s[2];
#pragma unroll
    for (int f = 0; f < 2; ++f) {
      s[f] = vzero8();
      int kb = min(kbase + f * 16, SEQL - 16);
#pragma unroll
      for (int c = 0; c < 2; ++c) {
        AF bf;
#pragma unroll
        for (int s = 0; s < 16; ++s) {
          int k = (s < 8) ? (half * 8 + s) : (16 + half * 8 + (s - 8));
          bf.v[s] = ktbase[(size_t)(c * 32 + k) * SEQL + kb + l15];
        }
        s[f] = wmma_bf16(qf[c].v, bf.v, s[f]);
      }
    }

    // ---- online softmax (row = r + 8*half, col = lane&15) ----
    float corr[8];
#pragma unroll
    for (int r = 0; r < 8; ++r) {
      int q = q0 + r + 8 * half;
      float s0 = s[0][r] * sscale;
      float s1 = s[1][r] * sscale;
      int k0 = kbase + l15;
      int k1 = kbase + 16 + l15;
      bool a0 = (k0 < SEQL) && ((q < NMEM) || (k0 <= q));
      bool a1 = (k1 < SEQL) && ((q < NMEM) || (k1 <= q));
      s0 = a0 ? s0 : -__builtin_inff();
      s1 = a1 ? s1 : -__builtin_inff();
      float mx = fmaxf(fmaxf(s0, s1), m[r]);
#pragma unroll
      for (int d = 1; d < 16; d <<= 1) mx = fmaxf(mx, __shfl_xor(mx, d, 32));
      float p0 = exp2f(s0 - mx);
      float p1 = exp2f(s1 - mx);
      float rs = p0 + p1;
#pragma unroll
      for (int d = 1; d < 16; d <<= 1) rs += __shfl_xor(rs, d, 32);
      float cr = exp2f(m[r] - mx);
      m[r] = mx;
      lsum[r] = lsum[r] * cr + rs;
      corr[r] = cr;
      pb[(r + 8 * half) * 40 + l15]      = (__bf16)p0;
      pb[(r + 8 * half) * 40 + 16 + l15] = (__bf16)p1;
    }

#pragma unroll
    for (int nf = 0; nf < 4; ++nf)
#pragma unroll
      for (int r = 0; r < 8; ++r) o[nf][r] *= corr[r];

    asm volatile("s_wait_dscnt 0" ::: "memory");

    AF pf;   // P in A-fragment layout (16x32)
    pf.q[0] = *(const uint4*)&pb[l15 * 40 + half * 8];
    pf.q[1] = *(const uint4*)&pb[l15 * 40 + 16 + half * 8];

    // ---- O += P * V ----
#pragma unroll
    for (int nf = 0; nf < 4; ++nf) {
      AF vf;
#pragma unroll
      for (int s = 0; s < 16; ++s) {
        int k = (s < 8) ? (half * 8 + s) : (16 + half * 8 + (s - 8));
        int vrow = min(kbase + k, SEQL - 1);
        vf.v[s] = vbase[(size_t)vrow * HD + nf * 16 + l15];
      }
      o[nf] = wmma_bf16(pf.v, vf.v, o[nf]);
    }
  }

  // ---- normalize + store y (B, L, C) bf16 ----
  const int b = bh >> 4, h = bh & (NH - 1);
#pragma unroll
  for (int r = 0; r < 8; ++r) {
    int q = q0 + r + 8 * half;
    if (q < SEQL) {
      float inv = 1.0f / lsum[r];
#pragma unroll
      for (int nf = 0; nf < 4; ++nf) {
        Y[((size_t)(b * SEQL + q)) * CDIM + h * HD + nf * 16 + l15] =
            (__bf16)(o[nf][r] * inv);
      }
    }
  }
}

// ---------------------------------------------------------------------------
// Proj GEMM: y(8224x1024 bf16) @ W_proj(1024x1024 bf16) -> out fp32
// Same async double-buffered structure as QKV GEMM.
// ---------------------------------------------------------------------------
__global__ __launch_bounds__(256) void proj_gemm_kernel(
    const __bf16* __restrict__ y, const __bf16* __restrict__ w,
    float* __restrict__ out) {
  __shared__ __bf16 As[2][TM * ASTR];
  __shared__ __bf16 Bs[2][TKK * BSTR];

  const int tid  = threadIdx.x;
  const int lane = tid & 31;
  const int wave = tid >> 5;
  const int wm   = wave >> 2;
  const int wn   = wave & 3;
  const int half = lane >> 4;
  const int l15  = lane & 15;
  const int mb   = blockIdx.x * TM;
  const int nb   = blockIdx.y * TN;

  const int ar0 = tid >> 2;
  const int ac0 = (tid & 3) * 8;
  const int br0 = tid >> 4;
  const int bc0 = (tid & 15) * 8;

  const __bf16* aA0 = y + (size_t)min(mb + ar0, NTOK - 1) * CDIM + ac0;
  const __bf16* aA1 = y + (size_t)min(mb + ar0 + 64, NTOK - 1) * CDIM + ac0;
  const __bf16* aB0 = w + (size_t)br0 * CDIM + nb + bc0;
  const __bf16* aB1 = w + (size_t)(br0 + 16) * CDIM + nb + bc0;

  auto issue = [&](int kk, int bufi) {
    async_cp16(aA0 + kk, &As[bufi][ar0 * ASTR + ac0]);
    async_cp16(aA1 + kk, &As[bufi][(ar0 + 64) * ASTR + ac0]);
    async_cp16(aB0 + (size_t)kk * CDIM, &Bs[bufi][br0 * BSTR + bc0]);
    async_cp16(aB1 + (size_t)kk * CDIM, &Bs[bufi][(br0 + 16) * BSTR + bc0]);
  };

  v8f acc[4][2];
#pragma unroll
  for (int mi = 0; mi < 4; ++mi)
#pragma unroll
    for (int ni = 0; ni < 2; ++ni) acc[mi][ni] = vzero8();

  issue(0, 0);
  for (int ks = 0; ks < NKSTEP; ++ks) {
    const int cur = ks & 1;
    if (ks + 1 < NKSTEP) {
      issue((ks + 1) * TKK, cur ^ 1);
      WAIT_ASYNC(4);
    } else {
      WAIT_ASYNC(0);
    }
    __syncthreads();

    AF a[4], b[2];
#pragma unroll
    for (int mi = 0; mi < 4; ++mi) {
      int row = wm * 64 + mi * 16 + l15;
      a[mi].q[0] = *(const uint4*)&As[cur][row * ASTR + half * 8];
      a[mi].q[1] = *(const uint4*)&As[cur][row * ASTR + 16 + half * 8];
    }
#pragma unroll
    for (int ni = 0; ni < 2; ++ni) {
      int col = wn * 32 + ni * 16;
#pragma unroll
      for (int s = 0; s < 16; ++s) {
        int k = (s < 8) ? (half * 8 + s) : (16 + half * 8 + (s - 8));
        b[ni].v[s] = Bs[cur][k * BSTR + col + l15];
      }
    }
#pragma unroll
    for (int mi = 0; mi < 4; ++mi)
#pragma unroll
      for (int ni = 0; ni < 2; ++ni)
        acc[mi][ni] = wmma_bf16(a[mi].v, b[ni].v, acc[mi][ni]);
    __syncthreads();
  }

#pragma unroll
  for (int mi = 0; mi < 4; ++mi)
#pragma unroll
    for (int ni = 0; ni < 2; ++ni)
#pragma unroll
      for (int r = 0; r < 8; ++r) {
        int grow = mb + wm * 64 + mi * 16 + r + 8 * half;
        int gcol = nb + wn * 32 + ni * 16 + l15;
        if (grow < NTOK) out[(size_t)grow * CDIM + gcol] = acc[mi][ni][r];
      }
}

// ---------------------------------------------------------------------------
extern "C" void kernel_launch(void* const* d_in, const int* in_sizes, int n_in,
                              void* d_out, int out_size, void* d_ws, size_t ws_size,
                              hipStream_t stream) {
  const float* x      = (const float*)d_in[0];
  const float* W_attn = (const float*)d_in[1];
  const float* W_proj = (const float*)d_in[2];
  // d_in[3] = n_head (16), d_in[4] = n_memory (8): fixed, compiled in.
  float* out = (float*)d_out;

  char* ws = (char*)d_ws;
  __bf16* xb      = (__bf16*)ws; ws += (size_t)NTOK * CDIM * 2;
  __bf16* wattn_b = (__bf16*)ws; ws += (size_t)CDIM * 3 * CDIM * 2;
  __bf16* wproj_b = (__bf16*)ws; ws += (size_t)CDIM * CDIM * 2;
  __bf16* Qb      = (__bf16*)ws; ws += (size_t)NTOK * CDIM * 2;
  __bf16* KTb     = (__bf16*)ws; ws += (size_t)NTOK * CDIM * 2;
  __bf16* Vb      = (__bf16*)ws; ws += (size_t)NTOK * CDIM * 2;
  __bf16* Yb      = (__bf16*)ws; ws += (size_t)NTOK * CDIM * 2;

  {
    int n4 = NTOK * CDIM / 4;
    cvt_kernel<<<(n4 + 255) / 256, 256, 0, stream>>>(x, xb, n4);
  }
  {
    int n4 = CDIM * 3 * CDIM / 4;
    cvt_kernel<<<(n4 + 255) / 256, 256, 0, stream>>>(W_attn, wattn_b, n4);
  }
  {
    int n4 = CDIM * CDIM / 4;
    cvt_kernel<<<(n4 + 255) / 256, 256, 0, stream>>>(W_proj, wproj_b, n4);
  }

  qkv_gemm_kernel<<<dim3((NTOK + TM - 1) / TM, 3 * CDIM / TN), 256, 0, stream>>>(
      xb, wattn_b, Qb, KTb, Vb);

  attn_kernel<<<dim3((SEQL + 63) / 64, BATCH * NH), 128, 0, stream>>>(
      Qb, KTb, Vb, Yb);

  proj_gemm_kernel<<<dim3((NTOK + TM - 1) / TM, CDIM / TN), 256, 0, stream>>>(
      Yb, wproj_b, out);
}
